// LSTMRegressor_420906795499
// MI455X (gfx1250) — matrix operand order repaired and checked by reading, not compiled, not run on hardware
//
#include <hip/hip_runtime.h>
#include <string.h>

// ---------------------------------------------------------------------------
// Types / constants
// ---------------------------------------------------------------------------
typedef __attribute__((ext_vector_type(16))) __bf16 v16bf;
typedef __attribute__((ext_vector_type(8)))  __bf16 v8bf;
typedef __attribute__((ext_vector_type(8)))  float  v8f;

#define B_    16
#define T_    512
#define H_    256
#define G_    1024   // 4*H
#define D2_   512    // 2*H
#define M_    8192   // B*T
#define DTOT_ 1664
#define OUT_  1000
#define HPAD  264    // padded LDS row stride (bf16 elems) to avoid bank conflicts

// ---------------------------------------------------------------------------
// Device helpers
// ---------------------------------------------------------------------------
__device__ __forceinline__ __bf16 f2bf(float f) {
  union { float f; unsigned u; } a; a.f = f;
  unsigned r = a.u + 0x7FFFu + ((a.u >> 16) & 1u);   // RNE
  union { unsigned short s; __bf16 b; } o; o.s = (unsigned short)(r >> 16);
  return o.b;
}
__device__ __forceinline__ float bf2f(__bf16 b) {
  union { unsigned short s; __bf16 b; } i; i.b = b;
  union { unsigned u; float f; } o; o.u = ((unsigned)i.s) << 16;
  return o.f;
}
__device__ __forceinline__ float sigm(float x) { return 1.0f / (1.0f + __expf(-x)); }
__device__ __forceinline__ float tanh_f(float x) {
  x = fminf(15.0f, fmaxf(-15.0f, x));
  float e = __expf(-2.0f * x);
  return (1.0f - e) / (1.0f + e);
}
__device__ __forceinline__ v16bf cat16(v8bf lo, v8bf hi) {
  return __builtin_shufflevector(lo, hi, 0,1,2,3,4,5,6,7,8,9,10,11,12,13,14,15);
}
// A fragment (16x32 bf16, M x K). Lane L: row = L&15; chunks at k+koffA, k+koffA+16
// with koffA = (L>=16 ? 8 : 0)  (ISA "16-bit A-Matrix 16x32" table).
__device__ __forceinline__ v16bf ldA(const __bf16* rowk, int koffA) {
  v8bf lo = *(const v8bf*)(rowk + koffA);
  v8bf hi = *(const v8bf*)(rowk + koffA + 16);
  return cat16(lo, hi);
}
// B fragment (32x16 bf16, K x N). Lane L: col = L&15; chunks at k+koffB/2, k+koffB/2+16
// with koffB = (L>=16 ? 16 : 0)  (same k order as the A fragment).
__device__ __forceinline__ v16bf ldB(const __bf16* rowk, int koffB) {
  v8bf lo = *(const v8bf*)(rowk + (koffB >> 1));
  v8bf hi = *(const v8bf*)(rowk + (koffB >> 1) + 16);
  return cat16(lo, hi);
}
__device__ __forceinline__ v8f wmma_bf16(v16bf a, v16bf b, v8f c) {
  return __builtin_amdgcn_wmma_f32_16x16x32_bf16(false, a, false, b, (short)0, c, false, false);
}

// ---------------------------------------------------------------------------
// Elementwise kernels
// ---------------------------------------------------------------------------
__global__ void k_cvt(const float* __restrict__ in, __bf16* __restrict__ out, int n) {
  int i = blockIdx.x * blockDim.x + threadIdx.x;
  if (i < n) out[i] = f2bf(in[i]);
}

// x (B, 1664, T) f32 -> xt (T, B, 1664) bf16
__global__ void k_transpose_x(const float* __restrict__ x, __bf16* __restrict__ xt, int n) {
  int i = blockIdx.x * blockDim.x + threadIdx.x;
  if (i >= n) return;
  int d  = i % DTOT_;
  int tb = i / DTOT_;        // = t*16 + b
  int b  = tb & 15;
  int t  = tb >> 4;
  xt[i] = f2bf(x[((size_t)b * DTOT_ + d) * T_ + t]);
}

__global__ void k_avg4(const __bf16* __restrict__ a, const __bf16* __restrict__ b,
                       const __bf16* __restrict__ c, const __bf16* __restrict__ d,
                       __bf16* __restrict__ o, int n) {
  int i = blockIdx.x * blockDim.x + threadIdx.x;
  if (i < n) o[i] = f2bf(0.25f * (bf2f(a[i]) + bf2f(b[i]) + bf2f(c[i]) + bf2f(d[i])));
}

// ---------------------------------------------------------------------------
// Input-projection GEMM: pre[m, n] = sum_k A[m,k]*W[n,k] + bih[n] + bhh[n]
// A bf16 (M_=8192 rows, row stride lda), W bf16 (1024 x K row-major),
// pre f32 (8192 x 1024). Block = 8 waves; wave computes 16(M) x 64(N).
// ---------------------------------------------------------------------------
__global__ __launch_bounds__(256) void k_gemm_pre(
    const __bf16* __restrict__ A, int lda, int K,
    const __bf16* __restrict__ W,
    const float* __restrict__ bih, const float* __restrict__ bhh,
    float* __restrict__ pre)
{
  const int tid = threadIdx.x, wave = tid >> 5, lane = tid & 31;
  const int nl = lane & 15;
  const int koffA = (lane >> 4) << 3;
  const int koffB = (lane >> 4) << 4;
  const int mref  = (lane >> 4) << 3;
  const int m0    = blockIdx.x << 4;
  const int nbase = (blockIdx.y << 9) + (wave << 6);

  v8f acc[4];
#pragma unroll
  for (int j = 0; j < 4; ++j) {
    const int n = nbase + j * 16 + nl;
    const float bs = bih[n] + bhh[n];
#pragma unroll
    for (int r = 0; r < 8; ++r) acc[j][r] = bs;
  }

  const __bf16* arow = A + (size_t)(m0 + nl) * lda;
  const int kt = K >> 5;
  for (int k = 0; k < kt; ++k) {
    const int k0 = k << 5;
    v16bf a = ldA(arow + k0, koffA);
#pragma unroll
    for (int j = 0; j < 4; ++j) {
      const __bf16* brow = W + (size_t)(nbase + j * 16 + nl) * K + k0;
      acc[j] = wmma_bf16(a, ldB(brow, koffB), acc[j]);
    }
  }

#pragma unroll
  for (int j = 0; j < 4; ++j) {
    const int n = nbase + j * 16 + nl;
#pragma unroll
    for (int r = 0; r < 8; ++r)
      pre[(size_t)(m0 + mref + r) * G_ + n] = acc[j][r];
  }
}

// ---------------------------------------------------------------------------
// Persistent LSTM scan. grid.x = 2 (0=forward, 1=reverse). 512 threads = 16 waves.
// Wave w owns hidden cols [16w, 16w+16) and all 4 gates for them, so gate math
// and cell state c stay in registers. h round-trips through LDS (bf16).
// pre is (T, B, 4H) f32; Whh bf16 (1024 x 256); out (T, B, 512) bf16, fwd cols
// 0..255, bwd cols 256..511 (bwd iterates t reversed).
// ---------------------------------------------------------------------------
__global__ __launch_bounds__(512) void k_lstm_scan(
    const float* __restrict__ preF, const float* __restrict__ preB,
    const __bf16* __restrict__ WhhF, const __bf16* __restrict__ WhhB,
    __bf16* __restrict__ out)
{
  const bool rev = (blockIdx.x != 0);
  const float*  pre = rev ? preB : preF;
  const __bf16* Whh = rev ? WhhB : WhhF;
  const int coloff  = rev ? H_ : 0;

  __shared__ __align__(16) __bf16 hbuf[16 * HPAD];

  const int tid = threadIdx.x, wave = tid >> 5, lane = tid & 31;
  const int nl = lane & 15;
  const int koffA = (lane >> 4) << 3;
  const int koffB = (lane >> 4) << 4;
  const int mref  = (lane >> 4) << 3;
  const int hcol  = (wave << 4) + nl;

  for (int i = tid; i < 16 * HPAD; i += 512) hbuf[i] = f2bf(0.0f);
  float cc[8];
#pragma unroll
  for (int r = 0; r < 8; ++r) cc[r] = 0.0f;
  __syncthreads();

  for (int ts = 0; ts < T_; ++ts) {
    const int t = rev ? (T_ - 1 - ts) : ts;
    const float* pt = pre + (size_t)t * B_ * G_;

    v8f acc[4];                                    // C := pre tile (gate q)
#pragma unroll
    for (int q = 0; q < 4; ++q) {
      const int n = q * H_ + hcol;
#pragma unroll
      for (int r = 0; r < 8; ++r) acc[q][r] = pt[(size_t)(mref + r) * G_ + n];
    }

    for (int k = 0; k < 8; ++k) {                  // K = 256 = 8 x 32
      const int k0 = k << 5;
      v16bf a = ldA(&hbuf[nl * HPAD + k0], koffA); // h from LDS (shared by 4 gates)
#pragma unroll
      for (int q = 0; q < 4; ++q) {
        const __bf16* brow = Whh + (size_t)(q * H_ + hcol) * H_ + k0;  // L2-resident
        acc[q] = wmma_bf16(a, ldB(brow, koffB), acc[q]);
      }
    }
    __syncthreads();                               // all waves done reading old h

#pragma unroll
    for (int r = 0; r < 8; ++r) {
      const float ig = sigm(acc[0][r]);
      const float fg = sigm(acc[1][r]);
      const float gg = tanh_f(acc[2][r]);
      const float og = sigm(acc[3][r]);
      const float cn = fg * cc[r] + ig * gg;
      cc[r] = cn;
      const float h = og * tanh_f(cn);
      const __bf16 hb = f2bf(h);
      hbuf[(mref + r) * HPAD + hcol] = hb;
      out[((size_t)t * B_ + mref + r) * D2_ + coloff + hcol] = hb;
    }
    __syncthreads();                               // new h visible for next step
  }
}

// ---------------------------------------------------------------------------
// Head GEMM with per-batch head selection.
// out[b, o, t] = sum_k post2[t,b,k] * headW[subj[b], o, k] + headb[subj[b], o]
// grid = (T/16, B). 512 threads = 16 waves; wave w covers o in [64w, 64w+64).
// ---------------------------------------------------------------------------
__global__ __launch_bounds__(512) void k_head(
    const __bf16* __restrict__ A, const __bf16* __restrict__ HW,
    const float* __restrict__ hb, const int* __restrict__ subject,
    float* __restrict__ out)
{
  const int b = blockIdx.y;
  int s = subject[b]; if (s == 5) s = 4; s -= 1;

  const int tid = threadIdx.x, wave = tid >> 5, lane = tid & 31;
  const int nl = lane & 15;
  const int koffA = (lane >> 4) << 3;
  const int koffB = (lane >> 4) << 4;
  const int mref  = (lane >> 4) << 3;
  const int m0    = blockIdx.x << 4;               // t tile
  const int nbase = wave << 6;

  const __bf16* HWs = HW + (size_t)s * OUT_ * D2_;
  const float*  hbs = hb + s * OUT_;

  v8f acc[4];
#pragma unroll
  for (int j = 0; j < 4; ++j) {
    const int n = nbase + j * 16 + nl;
    const float bb = (n < OUT_) ? hbs[n] : 0.0f;
#pragma unroll
    for (int r = 0; r < 8; ++r) acc[j][r] = bb;
  }

  const __bf16* arow = A + ((size_t)(m0 + nl) * B_ + b) * D2_;
  for (int k = 0; k < 16; ++k) {                   // K = 512
    const int k0 = k << 5;
    v16bf a = ldA(arow + k0, koffA);
#pragma unroll
    for (int j = 0; j < 4; ++j) {
      int n = nbase + j * 16 + nl; if (n >= OUT_) n = OUT_ - 1;   // clamp loads
      acc[j] = wmma_bf16(a, ldB(HWs + (size_t)n * D2_ + k0, koffB), acc[j]);
    }
  }

#pragma unroll
  for (int j = 0; j < 4; ++j) {
    const int n = nbase + j * 16 + nl;
    if (n < OUT_) {
#pragma unroll
      for (int r = 0; r < 8; ++r)
        out[((size_t)b * OUT_ + n) * T_ + m0 + mref + r] = acc[j][r];
    }
  }
}

// ---------------------------------------------------------------------------
// Host-side parameter mapping (handles insertion-order and jax-sorted pytrees)
// ---------------------------------------------------------------------------
static const int MODD_H[4] = {768, 512, 256, 128};

struct Mapped {
  const float* x = nullptr; const int* subject = nullptr;
  const float* headW = nullptr; const float* headb = nullptr;
  const float* Wih[20]; const float* Whh[20];
  const float* bih[20]; const float* bhh[20];
};

// dir index d: stacks 0..3 = modalities, 4 = post; d = (stack*2 + layer)*2 + (0 fwd / 1 bwd)
static int dirInDim(int d) {
  if (d >= 16) return 512;                 // post stack
  int m = d >> 2, l = (d >> 1) & 1;
  return l ? 512 : MODD_H[m];
}

static bool takeF(void* const* din, const int* sz, int n, int& i, long want, const float** dst) {
  if (i >= n || (long)sz[i] != want) return false;
  *dst = (const float*)din[i++]; return true;
}
static bool dirInsert(void* const* din, const int* sz, int n, int& i, Mapped& M, int d) {
  return takeF(din, sz, n, i, 1024L * dirInDim(d), &M.Wih[d])
      && takeF(din, sz, n, i, 262144L, &M.Whh[d])
      && takeF(din, sz, n, i, 1024L, &M.bih[d])
      && takeF(din, sz, n, i, 1024L, &M.bhh[d]);
}
static bool dirSorted(void* const* din, const int* sz, int n, int& i, Mapped& M, int d) {
  return takeF(din, sz, n, i, 262144L, &M.Whh[d])
      && takeF(din, sz, n, i, 1024L * dirInDim(d), &M.Wih[d])
      && takeF(din, sz, n, i, 1024L, &M.bhh[d])
      && takeF(din, sz, n, i, 1024L, &M.bih[d]);
}
static bool allDirsInsert(void* const* din, const int* sz, int n, int& i, Mapped& M) {
  for (int s = 0; s < 5; ++s)
    for (int l = 0; l < 2; ++l) {
      int base = (s * 2 + l) * 2;
      if (!dirInsert(din, sz, n, i, M, base)) return false;      // fwd
      if (!dirInsert(din, sz, n, i, M, base + 1)) return false;  // bwd
    }
  return true;
}
static bool allDirsSorted(void* const* din, const int* sz, int n, int& i, Mapped& M) {
  for (int s = 0; s < 5; ++s)
    for (int l = 0; l < 2; ++l) {
      int base = (s * 2 + l) * 2;
      if (!dirSorted(din, sz, n, i, M, base + 1)) return false;  // 'bwd' < 'fwd'
      if (!dirSorted(din, sz, n, i, M, base)) return false;
    }
  return true;
}
// A: pure insertion order: x; dirs(fwd,bwd; Wih,Whh,bih,bhh); headW; headb; subject
static bool tryOrderA(void* const* din, const int* sz, int n, Mapped& M) {
  int i = 0;
  if (!takeF(din, sz, n, i, 13631488L, &M.x)) return false;
  if (!allDirsInsert(din, sz, n, i, M)) return false;
  if (!takeF(din, sz, n, i, 2048000L, &M.headW)) return false;
  if (!takeF(din, sz, n, i, 4000L, &M.headb)) return false;
  if (i >= n || sz[i] != 16) return false; M.subject = (const int*)din[i++];
  return i == n;
}
// C: top-level insertion, params jax-sorted: x; headW; headb; dirs(sorted); subject
static bool tryOrderC(void* const* din, const int* sz, int n, Mapped& M) {
  int i = 0;
  if (!takeF(din, sz, n, i, 13631488L, &M.x)) return false;
  if (!takeF(din, sz, n, i, 2048000L, &M.headW)) return false;
  if (!takeF(din, sz, n, i, 4000L, &M.headb)) return false;
  if (!allDirsSorted(din, sz, n, i, M)) return false;
  if (i >= n || sz[i] != 16) return false; M.subject = (const int*)din[i++];
  return i == n;
}
// B: fully jax-sorted top-level: params(headW,headb,mods,post); subject; x
static bool tryOrderB(void* const* din, const int* sz, int n, Mapped& M) {
  int i = 0;
  if (!takeF(din, sz, n, i, 2048000L, &M.headW)) return false;
  if (!takeF(din, sz, n, i, 4000L, &M.headb)) return false;
  if (!allDirsSorted(din, sz, n, i, M)) return false;
  if (i >= n || sz[i] != 16) return false; M.subject = (const int*)din[i++];
  if (!takeF(din, sz, n, i, 13631488L, &M.x)) return false;
  return i == n;
}
static void forceOrderA(void* const* din, int n, Mapped& M) {
  int i = 0;
  M.x = (const float*)din[i++];
  for (int s = 0; s < 5; ++s)
    for (int l = 0; l < 2; ++l)
      for (int dd = 0; dd < 2; ++dd) {
        int d = (s * 2 + l) * 2 + dd;
        if (i + 3 >= n) return;
        M.Wih[d] = (const float*)din[i++]; M.Whh[d] = (const float*)din[i++];
        M.bih[d] = (const float*)din[i++]; M.bhh[d] = (const float*)din[i++];
      }
  if (i < n) M.headW = (const float*)din[i++];
  if (i < n) M.headb = (const float*)din[i++];
  if (i < n) M.subject = (const int*)din[i++];
}

// ---------------------------------------------------------------------------
// Launch orchestration
// ---------------------------------------------------------------------------
extern "C" void kernel_launch(void* const* d_in, const int* in_sizes, int n_in,
                              void* d_out, int out_size, void* d_ws, size_t ws_size,
                              hipStream_t stream)
{
  (void)out_size; (void)ws_size;
  Mapped M;
  if (!tryOrderA(d_in, in_sizes, n_in, M))
    if (!tryOrderC(d_in, in_sizes, n_in, M))
      if (!tryOrderB(d_in, in_sizes, n_in, M))
        forceOrderA(d_in, n_in, M);

  // ---- workspace carve (bytes, 256B aligned) ----
  size_t off = 0;
  auto carve = [&](size_t bytes) -> void* {
    void* p = (char*)d_ws + off;
    off = (off + bytes + 255) & ~(size_t)255;
    return p;
  };
  __bf16* xt   = (__bf16*)carve((size_t)M_ * DTOT_ * 2);   // (T,B,1664) bf16
  float*  preF = (float*) carve((size_t)M_ * G_ * 4);      // (T,B,4H) f32
  float*  preB = (float*) carve((size_t)M_ * G_ * 4);
  __bf16* stk  = (__bf16*)carve((size_t)M_ * D2_ * 2);     // layer-1 bidir out
  __bf16* modo[4];
  for (int m = 0; m < 4; ++m) modo[m] = (__bf16*)carve((size_t)M_ * D2_ * 2);
  __bf16* avgb  = (__bf16*)carve((size_t)M_ * D2_ * 2);
  __bf16* post1 = (__bf16*)carve((size_t)M_ * D2_ * 2);
  __bf16* post2 = (__bf16*)carve((size_t)M_ * D2_ * 2);

  size_t woffI[20], woffH[20], wtot = 0;
  for (int d = 0; d < 20; ++d) {
    woffI[d] = wtot; wtot += (size_t)1024 * dirInDim(d);
    woffH[d] = wtot; wtot += 262144;
  }
  size_t hwoff = wtot; wtot += 2048000;
  __bf16* wslab = (__bf16*)carve(wtot * 2);

  auto cvt = [&](const float* s, __bf16* dst, int n) {
    k_cvt<<<dim3((unsigned)((n + 255) / 256)), 256, 0, stream>>>(s, dst, n);
  };

  // transpose + convert inputs / weights to bf16 (re-done every call: deterministic)
  {
    int n = (int)((size_t)M_ * DTOT_);
    k_transpose_x<<<dim3((unsigned)((n + 255) / 256)), 256, 0, stream>>>(M.x, xt, n);
  }
  for (int d = 0; d < 20; ++d) {
    cvt(M.Wih[d], wslab + woffI[d], 1024 * dirInDim(d));
    cvt(M.Whh[d], wslab + woffH[d], 262144);
  }
  cvt(M.headW, wslab + hwoff, 2048000);

  const dim3 gg(512, 2);
  int xoff = 0;
  for (int m = 0; m < 4; ++m) {
    const int K = MODD_H[m];
    const int d0 = (m * 2 + 0) * 2;
    k_gemm_pre<<<gg, 256, 0, stream>>>(xt + xoff, DTOT_, K, wslab + woffI[d0],     M.bih[d0],     M.bhh[d0],     preF);
    k_gemm_pre<<<gg, 256, 0, stream>>>(xt + xoff, DTOT_, K, wslab + woffI[d0 + 1], M.bih[d0 + 1], M.bhh[d0 + 1], preB);
    k_lstm_scan<<<2, 512, 0, stream>>>(preF, preB, wslab + woffH[d0], wslab + woffH[d0 + 1], stk);
    const int d1 = (m * 2 + 1) * 2;
    k_gemm_pre<<<gg, 256, 0, stream>>>(stk, D2_, D2_, wslab + woffI[d1],     M.bih[d1],     M.bhh[d1],     preF);
    k_gemm_pre<<<gg, 256, 0, stream>>>(stk, D2_, D2_, wslab + woffI[d1 + 1], M.bih[d1 + 1], M.bhh[d1 + 1], preB);
    k_lstm_scan<<<2, 512, 0, stream>>>(preF, preB, wslab + woffH[d1], wslab + woffH[d1 + 1], modo[m]);
    xoff += K;
  }
  {
    int n = (int)((size_t)M_ * D2_);
    k_avg4<<<dim3((unsigned)((n + 255) / 256)), 256, 0, stream>>>(modo[0], modo[1], modo[2], modo[3], avgb, n);
  }
  {
    const int d0 = 16;
    k_gemm_pre<<<gg, 256, 0, stream>>>(avgb, D2_, D2_, wslab + woffI[d0],     M.bih[d0],     M.bhh[d0],     preF);
    k_gemm_pre<<<gg, 256, 0, stream>>>(avgb, D2_, D2_, wslab + woffI[d0 + 1], M.bih[d0 + 1], M.bhh[d0 + 1], preB);
    k_lstm_scan<<<2, 512, 0, stream>>>(preF, preB, wslab + woffH[d0], wslab + woffH[d0 + 1], post1);
    const int d1 = 18;
    k_gemm_pre<<<gg, 256, 0, stream>>>(post1, D2_, D2_, wslab + woffI[d1],     M.bih[d1],     M.bhh[d1],     preF);
    k_gemm_pre<<<gg, 256, 0, stream>>>(post1, D2_, D2_, wslab + woffI[d1 + 1], M.bih[d1 + 1], M.bhh[d1 + 1], preB);
    k_lstm_scan<<<2, 512, 0, stream>>>(preF, preB, wslab + woffH[d1], wslab + woffH[d1 + 1], post2);
  }
  k_head<<<dim3(32, 16), 512, 0, stream>>>(post2, wslab + hwoff, M.headb, M.subject, (float*)d_out);
}
